// FusionDeepONet_24249385353577
// MI455X (gfx1250) — compile-verified, matrix operand order checked by reading, not run on hardware
//
#include <hip/hip_runtime.h>
#include <math.h>

#define COORD_DIM 3
#define PARAM_DIM 8
#define H 128
#define L 5
#define O 5
#define B_SZ 16
#define N_PTS 50000
#define TILE_M 128
#define N_TILES ((N_PTS + TILE_M - 1) / TILE_M)   // 391

typedef _Float16 v16h __attribute__((ext_vector_type(16)));
typedef _Float16 v8h  __attribute__((ext_vector_type(8)));
typedef float    v8f  __attribute__((ext_vector_type(8)));

union V16 { v16h v; v8h h[2]; };

// CDNA5 hardware tanh (V_TANH_F32, TRANS class — co-executes with XDL WMMA)
#if __has_builtin(__builtin_amdgcn_tanh_f32)
__device__ __forceinline__ float fast_tanh(float x) {
    return __builtin_amdgcn_tanh_f32(x);
}
#elif __has_builtin(__builtin_amdgcn_tanhf)
__device__ __forceinline__ float fast_tanh(float x) {
    return __builtin_amdgcn_tanhf(x);
}
#else
__device__ __forceinline__ float fast_tanh(float x) { return tanhf(x); }
#endif

// ---- workspace layout -------------------------------------------------------
// S  : float   [L][B][H]      (cumsum FiLM vectors)      5*16*128 floats
// ZL : float   [B][O*H]       (branch output)            16*640  floats
// WT : _Float16[5][H][H]      (transposed f16 weights: WhT[0..3], WfT)
#define WS_S_ELEMS  (L * B_SZ * H)
#define WS_ZL_ELEMS (B_SZ * O * H)
#define WS_WT_BYTES ((size_t)(WS_S_ELEMS + WS_ZL_ELEMS) * sizeof(float))

// ---- prep kernel 1: branch MLP, cumsums S_i, ZL -----------------------------
__global__ __launch_bounds__(128) void branch_prep(
    const float* __restrict__ params, const float* __restrict__ W0,
    const float* __restrict__ b0, const float* __restrict__ Wh,
    const float* __restrict__ bh, const float* __restrict__ alpha,
    const float* __restrict__ Wf, const float* __restrict__ bf,
    float* __restrict__ S, float* __restrict__ ZL)
{
    __shared__ float hbuf[2][B_SZ][H];
    const int t = threadIdx.x;                    // output-feature index
    float scum[B_SZ];

    // layer 0: params(8) -> H, rowdy tanh
    for (int b = 0; b < B_SZ; ++b) {
        float acc = b0[t];
        #pragma unroll
        for (int p = 0; p < PARAM_DIM; ++p)
            acc += params[b * PARAM_DIM + p] * W0[p * H + t];
        float h = alpha[t] * fast_tanh(acc);
        hbuf[0][b][t] = h;
        scum[b] = h;
        S[(0 * B_SZ + b) * H + t] = h;
    }
    __syncthreads();

    int cur = 0;
    for (int i = 0; i < L - 1; ++i) {
        int nxt = cur ^ 1;
        for (int b = 0; b < B_SZ; ++b) {
            float acc = bh[i * H + t];
            for (int k = 0; k < H; ++k)
                acc += hbuf[cur][b][k] * Wh[(i * H + k) * H + t];
            float h = alpha[(i + 1) * H + t] * fast_tanh(acc);
            hbuf[nxt][b][t] = h;
            scum[b] += h;
            S[((i + 1) * B_SZ + b) * H + t] = scum[b];
        }
        __syncthreads();
        cur = nxt;
    }

    // branch_out = h4 @ Wf + bf  -> ZL[b][o*H+h]
    for (int b = 0; b < B_SZ; ++b)
        for (int o = 0; o < O; ++o) {
            int j = o * H + t;
            float acc = bf[j];
            for (int k = 0; k < H; ++k)
                acc += hbuf[cur][b][k] * Wf[k * (O * H) + j];
            ZL[b * (O * H) + j] = acc;
        }
}

// ---- prep kernel 2: transpose trunk weights to f16: WT[mat][n][k] ----------
__global__ __launch_bounds__(256) void wt_prep(
    const float* __restrict__ Wh, const float* __restrict__ Wf,
    _Float16* __restrict__ WT)
{
    const int mat = blockIdx.x;                   // 0..3 = WhT, 4 = WfT
    const float* src = (mat < 4) ? (Wh + mat * H * H) : Wf;
    _Float16* dst = WT + mat * H * H;
    for (int i = threadIdx.x; i < H * H; i += 256) {
        int n = i >> 7, k = i & 127;
        dst[i] = (_Float16)src[k * H + n];
    }
}

// ---- main kernel: trunk with FiLM + WMMA matmuls + output contraction ------
__global__ __launch_bounds__(256) void trunk_main(
    const float* __restrict__ coords,
    const float* __restrict__ tW0, const float* __restrict__ tb0,
    const float* __restrict__ tbh, const float* __restrict__ talpha,
    const float* __restrict__ tbf,
    const float* __restrict__ S, const float* __restrict__ ZL,
    const _Float16* __restrict__ WT,
    float* __restrict__ out)
{
    // activation tile, rows padded to 136 halves (272B -> 4-bank row skew)
    __shared__ _Float16 X[TILE_M][H + 8];

    const int b    = blockIdx.x / N_TILES;
    const int tile = blockIdx.x % N_TILES;
    const int p0   = tile * TILE_M;

    const int tid  = threadIdx.x;
    const int wave = tid >> 5;
    const int lane = tid & 31;
    const int lrow = lane & 15;   // A-fragment row / D-fragment N index
    const int hi   = lane >> 4;   // lane-half selects K/M sub-block
    const int m0   = wave * 16;   // this wave's private 16 rows

    // ---- layer 0: coords(3) -> H, FiLM*S0, rowdy tanh (VALU; K=3 too small
    //      for WMMA). Each wave writes only its own 16 rows -> no barriers.
    {
        float w00[4], w01[4], w02[4], bb[4], ss[4], aa[4];
        #pragma unroll
        for (int j = 0; j < 4; ++j) {
            int n = lane + 32 * j;
            w00[j] = tW0[0 * H + n];
            w01[j] = tW0[1 * H + n];
            w02[j] = tW0[2 * H + n];
            bb[j]  = tb0[n];
            ss[j]  = S[(0 * B_SZ + b) * H + n];
            aa[j]  = talpha[0 * H + n];
        }
        for (int m = 0; m < 16; ++m) {
            int p = p0 + m0 + m;
            if (p >= N_PTS) p = N_PTS - 1;        // pad rows: computed, never stored
            const float* c = coords + ((size_t)b * N_PTS + p) * COORD_DIM;
            float c0 = c[0], c1 = c[1], c2 = c[2];
            #pragma unroll
            for (int j = 0; j < 4; ++j) {
                float v = c0 * w00[j] + c1 * w01[j] + c2 * w02[j] + bb[j];
                v = aa[j] * fast_tanh(v * ss[j]);
                X[m0 + m][lane + 32 * j] = (_Float16)v;
            }
        }
    }

    // ---- layers 1..4 (Wh, FiLM+tanh) and layer 5 (Wf, bias only) ----------
    for (int layer = 1; layer <= L; ++layer) {
        const bool isFinal = (layer == L);
        const _Float16* Wl = WT + (layer - 1) * H * H;  // WhT[layer-1] or WfT

        // A fragments: 16x32 f16 per k-step, from this wave's own LDS rows.
        // Lane layout (ISA 7.12.2): lanes 0-15 hold K kb..kb+7 & kb+16..kb+23.
        V16 afrag[4];
        #pragma unroll
        for (int ks = 0; ks < 4; ++ks) {
            int kb = ks * 32 + hi * 8;
            const _Float16* xr = &X[m0 + lrow][0];
            afrag[ks].h[0] = *(const v8h*)(xr + kb);
            afrag[ks].h[1] = *(const v8h*)(xr + kb + 16);
        }

        #pragma unroll
        for (int nt = 0; nt < 8; ++nt) {
            const int n = nt * 16 + lrow;         // output column for this lane
            v8f acc = {};
            #pragma unroll
            for (int ks = 0; ks < 4; ++ks) {
                int kb = ks * 32 + hi * 8;
                V16 bfrag;                        // B from transposed weights
                const _Float16* wr = Wl + n * H;
                bfrag.h[0] = *(const v8h*)(wr + kb);
                bfrag.h[1] = *(const v8h*)(wr + kb + 16);
                acc = __builtin_amdgcn_wmma_f32_16x16x32_f16(
                        false, afrag[ks].v, false, bfrag.v,
                        (short)0, acc, false, false);
            }
            // D layout: vgpr r -> M = hi*8 + r, N = lrow. Bias/S/alpha are
            // per-N => per-lane scalars.
            if (!isFinal) {
                float bias = tbh[(layer - 1) * H + n];
                float ssc  = S[(layer * B_SZ + b) * H + n];
                float aal  = talpha[layer * H + n];
                #pragma unroll
                for (int r = 0; r < 8; ++r) {
                    float v = acc[r] + bias;
                    v = aal * fast_tanh(v * ssc);
                    X[m0 + hi * 8 + r][n] = (_Float16)v;
                }
            } else {
                float bias = tbf[n];
                #pragma unroll
                for (int r = 0; r < 8; ++r)
                    X[m0 + hi * 8 + r][n] = (_Float16)(acc[r] + bias);
            }
        }
        // no barrier: each wave reads/writes only rows [m0, m0+16);
        // per-wave LDS ops are in-order (DScnt).
    }

    // ---- output: out[b,p,o] = sum_h ZL[b,o,h] * YL[p,h] (O=5, VALU dots) ---
    {
        const float* Zb = ZL + (size_t)b * O * H;
        for (int idx = lane; idx < 16 * O; idx += 32) {
            int mm = idx / O;
            int o  = idx % O;
            int p  = p0 + m0 + mm;
            if (p < N_PTS) {
                const _Float16* y = &X[m0 + mm][0];
                const float* z = Zb + o * H;
                float acc = 0.f;
                #pragma unroll
                for (int h8 = 0; h8 < 16; ++h8) {
                    v8h yv = *(const v8h*)(y + h8 * 8);
                    #pragma unroll
                    for (int j = 0; j < 8; ++j)
                        acc += (float)yv[j] * z[h8 * 8 + j];
                }
                out[((size_t)b * N_PTS + p) * O + o] = acc;
            }
        }
    }
}

extern "C" void kernel_launch(void* const* d_in, const int* in_sizes, int n_in,
                              void* d_out, int out_size, void* d_ws, size_t ws_size,
                              hipStream_t stream) {
    (void)in_sizes; (void)n_in; (void)out_size; (void)ws_size;

    const float* coords = (const float*)d_in[0];
    const float* params = (const float*)d_in[1];
    const float* bW0    = (const float*)d_in[2];
    const float* bb0    = (const float*)d_in[3];
    const float* bWh    = (const float*)d_in[4];
    const float* bbh    = (const float*)d_in[5];
    const float* balpha = (const float*)d_in[6];
    const float* bWf    = (const float*)d_in[7];
    const float* bbf    = (const float*)d_in[8];
    const float* tW0    = (const float*)d_in[9];
    const float* tb0    = (const float*)d_in[10];
    const float* tWh    = (const float*)d_in[11];
    const float* tbh    = (const float*)d_in[12];
    const float* talpha = (const float*)d_in[13];
    const float* tWf    = (const float*)d_in[14];
    const float* tbf    = (const float*)d_in[15];

    float*    S  = (float*)d_ws;
    float*    ZL = S + WS_S_ELEMS;
    _Float16* WT = (_Float16*)((char*)d_ws + WS_WT_BYTES);

    branch_prep<<<1, 128, 0, stream>>>(params, bW0, bb0, bWh, bbh,
                                       balpha, bWf, bbf, S, ZL);
    wt_prep<<<5, 256, 0, stream>>>(tWh, tWf, WT);
    trunk_main<<<B_SZ * N_TILES, 256, 0, stream>>>(
        coords, tW0, tb0, tbh, talpha, tbf, S, ZL, WT, (float*)d_out);
}
